// DARTSLayer_82514911690830
// MI455X (gfx1250) — compile-verified, matrix operand order checked by reading, not run on hardware
//
#include <hip/hip_runtime.h>

// ---------------------------------------------------------------------------
// DARTS mixed-op layer as a skinny GEMM on CDNA5 WMMA.
//   out[b,j] = sum_{i,k} prims(x)[b,i,k] * (softmax(alphas)*coeffs)[i,j,k]
// Fold kappa = i*8+k  ->  out = P[65536x512] @ W[512x64], f16 in / f32 acc.
// Roofline: 4.3 GFLOP vs 32 MB HBM traffic (x in + out once, W resident in
// LDS) -> ~1.4 us memory floor; f16 WMMA compute is the same order, so the
// design keeps both pipes busy: trans-ops build A while WMMA consumes B/LDS.
// ---------------------------------------------------------------------------

typedef __attribute__((ext_vector_type(16))) _Float16 v16h;
typedef __attribute__((ext_vector_type(8)))  float    v8f;

#define N_IN   64
#define N_OUT  64
#define N_OPS  8
#define KDIM   (N_IN * N_OPS)      // 512
#define BATCH  65536
#define KCHUNKS (KDIM / 32)        // 16 chunks of K=32 per wmma
#define NTILES  (N_OUT / 16)       // 4 tiles of N=16
// W fragment buffer: [kchunk][ntile][lane][16 halves] = 16*4*32*16 halves = 64KB
#define WFRAG_HALVES (KCHUNKS * NTILES * 32 * 16)

// exact-domain scaling constants for raw hardware transcendentals
#define LOG2E    1.44269504088896340736f   // exp(x)  = exp2(x * LOG2E)
#define LN2      0.69314718055994530942f   // ln(x)   = log2(x) * LN2
#define INV_2PI  0.15915494309189533577f   // sin(x)  = v_sin(x * INV_2PI)

// ---------------------------------------------------------------------------
// Kernel 1: fold softmax(alphas)*coeffs into f16 W, stored directly in the
// wave32 B-fragment layout for V_WMMA_F32_16X16X32_F16 (ISA 7.12.2):
//   lane 0-15  = column N (=lane),  halves 0-7 -> K 0-7,  halves 8-15 -> K 16-23
//   lane 16-31 = column N (=lane-16), halves 0-7 -> K 8-15, halves 8-15 -> K 24-31
// One thread per (kchunk, ntile, lane) slot -> one 32-byte store.
// ---------------------------------------------------------------------------
__global__ __launch_bounds__(256) void darts_prep_w(
    const float* __restrict__ alphas,   // [I=64][J=64][K=8]
    const float* __restrict__ coeffs,   // [I=64][J=64][K=8]
    _Float16* __restrict__ wfrag)       // [WFRAG_HALVES]
{
    int t = blockIdx.x * blockDim.x + threadIdx.x;     // 0 .. 2047
    if (t >= KCHUNKS * NTILES * 32) return;
    int lane = t & 31;
    int nt   = (t >> 5) & (NTILES - 1);
    int kc   = t >> 7;
    int n    = nt * 16 + (lane & 15);                  // output column j

    v16h out;
    #pragma unroll
    for (int h = 0; h < 16; ++h) {
        // half h -> K offset inside this 32-wide chunk (B-frag layout above)
        int koff = h + ((h & 8) ? 8 : 0) + ((lane & 16) ? 8 : 0);
        int kg = kc * 32 + koff;                       // global kappa
        int i  = kg >> 3;                              // input feature
        int k  = kg & 7;                               // primitive op
        const float* a = alphas + (i * N_OUT + n) * N_OPS;
        float m = a[0];
        #pragma unroll
        for (int q = 1; q < N_OPS; ++q) m = fmaxf(m, a[q]);
        float s = 0.0f;
        #pragma unroll
        for (int q = 0; q < N_OPS; ++q) s += __expf(a[q] - m);
        float gate = __expf(a[k] - m) / s;
        float w = gate * coeffs[(i * N_OUT + n) * N_OPS + k];
        out[h] = (_Float16)w;
    }
    *reinterpret_cast<v16h*>(wfrag + t * 16) = out;
}

// ---------------------------------------------------------------------------
// Kernel 2: the GEMM. 256 threads = 8 waves per block; each wave owns a
// 16-row M tile x all 64 columns (4 x v8f accumulators). W lives in LDS.
// A fragment (16x32 f16, ISA 7.12.2): lane = row M; lanes 0-15 hold
// K {0-7,16-23} -> inputs i0, i0+2; lanes 16-31 hold K {8-15,24-31}
// -> inputs i0+1, i0+3. Each lane computes 2x8 primitives per chunk using
// raw hardware trans-ops (domain (0.5,1.5) -> no denorm/overflow handling
// needed, and results are truncated to f16 anyway).
// ---------------------------------------------------------------------------
__global__ __launch_bounds__(256) void darts_gemm(
    const float*    __restrict__ x,      // [BATCH][64]
    const _Float16* __restrict__ wfrag,  // [WFRAG_HALVES]
    float*          __restrict__ out)    // [BATCH][64]
{
    __shared__ uint4 lw[WFRAG_HALVES / 8];             // 64 KB of f16 W frags

    // cooperative 16B-granularity copy of W into LDS
    const uint4* src = reinterpret_cast<const uint4*>(wfrag);
    for (int idx = threadIdx.x; idx < WFRAG_HALVES / 8; idx += 256)
        lw[idx] = src[idx];
    __syncthreads();

    const int lane = threadIdx.x & 31;
    const int wave = threadIdx.x >> 5;
    const int m0   = (blockIdx.x * 8 + wave) * 16;     // first batch row of tile
    const int row  = m0 + (lane & 15);

    v8f acc0 = {}, acc1 = {}, acc2 = {}, acc3 = {};

    #pragma unroll 4
    for (int kc = 0; kc < KCHUNKS; ++kc) {
        const int i0 = kc * 4;                         // 4 input features / chunk
        float4 xq = *reinterpret_cast<const float4*>(x + row * N_IN + i0);
        float xa = (lane & 16) ? xq.y : xq.x;          // input i0 (+1 for hi lanes)
        float xb = (lane & 16) ? xq.w : xq.z;          // input i0+2 (+1 for hi lanes)

        v16h a;
        float xa2 = xa * xa;
        float xb2 = xb * xb;
        a[0]  = (_Float16)0.0f;                                   // none
        a[1]  = (_Float16)xa;                                     // linear
        a[2]  = (_Float16)xa2;                                    // x^2
        a[3]  = (_Float16)(xa2 * xa);                             // x^3
        a[4]  = (_Float16)__builtin_amdgcn_exp2f(xa * LOG2E);     // exp
        a[5]  = (_Float16)(__builtin_amdgcn_logf(xa) * LN2);      // ln
        a[6]  = (_Float16)__builtin_amdgcn_rcpf(xa);              // 1/x
        a[7]  = (_Float16)__builtin_amdgcn_sinf(xa * INV_2PI);    // sin
        a[8]  = (_Float16)0.0f;
        a[9]  = (_Float16)xb;
        a[10] = (_Float16)xb2;
        a[11] = (_Float16)(xb2 * xb);
        a[12] = (_Float16)__builtin_amdgcn_exp2f(xb * LOG2E);
        a[13] = (_Float16)(__builtin_amdgcn_logf(xb) * LN2);
        a[14] = (_Float16)__builtin_amdgcn_rcpf(xb);
        a[15] = (_Float16)__builtin_amdgcn_sinf(xb * INV_2PI);

        union { v16h h; uint4 q[2]; } b;
        int base = (kc * NTILES * 32 + lane) * 2;      // uint4 index into lw

        b.q[0] = lw[base +  0]; b.q[1] = lw[base +  1];
        acc0 = __builtin_amdgcn_wmma_f32_16x16x32_f16(false, a, false, b.h,
                                                      (short)0, acc0, false, false);
        b.q[0] = lw[base + 64]; b.q[1] = lw[base + 65];
        acc1 = __builtin_amdgcn_wmma_f32_16x16x32_f16(false, a, false, b.h,
                                                      (short)0, acc1, false, false);
        b.q[0] = lw[base + 128]; b.q[1] = lw[base + 129];
        acc2 = __builtin_amdgcn_wmma_f32_16x16x32_f16(false, a, false, b.h,
                                                      (short)0, acc2, false, false);
        b.q[0] = lw[base + 192]; b.q[1] = lw[base + 193];
        acc3 = __builtin_amdgcn_wmma_f32_16x16x32_f16(false, a, false, b.h,
                                                      (short)0, acc3, false, false);
    }

    // C/D layout (ISA 7.12.2): VGPR r holds M=r (lanes 0-15) / M=r+8 (lanes 16-31),
    // N = lane&15.
    const int col0 = lane & 15;
    const int rb   = m0 + ((lane >> 4) << 3);
    #pragma unroll
    for (int r = 0; r < 8; ++r) {
        float* o = out + (rb + r) * N_OUT + col0;
        o[0]  = acc0[r];
        o[16] = acc1[r];
        o[32] = acc2[r];
        o[48] = acc3[r];
    }
}

// ---------------------------------------------------------------------------
extern "C" void kernel_launch(void* const* d_in, const int* in_sizes, int n_in,
                              void* d_out, int out_size, void* d_ws, size_t ws_size,
                              hipStream_t stream) {
    const float* x      = (const float*)d_in[0];   // [65536*64]
    const float* alphas = (const float*)d_in[1];   // [64*64*8]
    const float* coeffs = (const float*)d_in[2];   // [64*64*8]
    float*       out    = (float*)d_out;           // [65536*64]
    _Float16*    wfrag  = (_Float16*)d_ws;         // 64 KB scratch

    // 2048 slots -> 8 blocks of 256
    darts_prep_w<<<8, 256, 0, stream>>>(alphas, coeffs, wfrag);

    // 128 batch rows per block -> 512 blocks
    darts_gemm<<<BATCH / 128, 256, 0, stream>>>(x, wfrag, out);
}
